// GatedGNN_89154931130413
// MI455X (gfx1250) — matrix operand order repaired and checked by reading, not compiled, not run on hardware
//
#include <hip/hip_runtime.h>

// GatedGNN fused CDNA5 implementation (bf16 WMMA + TDM + async-LDS path).
// B=8, N=2048, D=256.

#define BATCH 8
#define NNODE 2048
#define DIM   256
#define BN    (BATCH * NNODE)      // 16384
#define NEGV  (-2.147483648e9f)

typedef __attribute__((ext_vector_type(16))) __bf16 v16bf;
typedef __attribute__((ext_vector_type(8)))  float  v8f;
typedef unsigned int u32x4 __attribute__((ext_vector_type(4)));
typedef int          i32x4 __attribute__((ext_vector_type(4)));
typedef int          i32x8 __attribute__((ext_vector_type(8)));

union BfVec {
    v16bf v;
    unsigned short s[16];
    uint4 q[2];
};
union F8 {
    v8f v;
    float f[8];
};

__device__ __forceinline__ unsigned short f2bf(float f) {
    union { float f; unsigned int u; } cv; cv.f = f;
    unsigned int u = cv.u;
    unsigned int rnd = 0x7FFFu + ((u >> 16) & 1u);   // round-to-nearest-even
    return (unsigned short)((u + rnd) >> 16);
}

// Low 32 bits of the LDS-aperture flat address == LDS byte offset.
__device__ __forceinline__ unsigned lds_off(const void* p) {
    return (unsigned)(uintptr_t)p;
}

// Async global->LDS 16-byte copy (per-lane, GV mode), tracked by ASYNCcnt.
__device__ __forceinline__ void async_g2l_b128(unsigned lds_byte_off, const void* g) {
    asm volatile("global_load_async_to_lds_b128 %0, %1, off"
                 :: "v"(lds_byte_off), "v"((unsigned long long)(uintptr_t)g)
                 : "memory");
}

// A operand, 16x32 bf16 tile: rows M in [0,16), K chunk kc (32 wide).
__device__ __forceinline__ v16bf load_a16(const unsigned short* base, int stride,
                                          int lane, int kc) {
    BfVec u;
    int m  = lane & 15;
    int hf = lane >> 4;
    int k0 = kc * 32 + hf * 16;
    int k1 = kc * 32 + hf * 16 + 8;
    __builtin_memcpy(&u.q[0], base + (size_t)m * stride + k0, 16);
    __builtin_memcpy(&u.q[1], base + (size_t)m * stride + k1, 16);
    return u.v;
}

// B operand, 32x16 bf16 tile stored as [ncol][K-contig] rows.
__device__ __forceinline__ v16bf load_b16(const unsigned short* colbase, int stride,
                                          int lane, int kc) {
    BfVec u;
    int n  = lane & 15;
    int hf = lane >> 4;
    int k0 = kc * 32 + hf * 16;
    __builtin_memcpy(&u.q[0], colbase + (size_t)n * stride + k0, 16);
    __builtin_memcpy(&u.q[1], colbase + (size_t)n * stride + k0 + 8, 16);
    return u.v;
}

// ---------------------------------------------------------------------------
// K1: row-normalize x_topic -> ybf (bf16 unit rows), convert x -> xbf (bf16).
// ---------------------------------------------------------------------------
__global__ void k1_prep(const float* __restrict__ x, const float* __restrict__ xt,
                        unsigned short* __restrict__ xbf,
                        unsigned short* __restrict__ ybf) {
    int wv = threadIdx.x >> 5, lane = threadIdx.x & 31;
    int row = blockIdx.x * 8 + wv;
    const float* xr = xt + (size_t)row * DIM + lane * 8;
    float v[8];
    float ss = 0.f;
#pragma unroll
    for (int i = 0; i < 8; ++i) { v[i] = xr[i]; ss += v[i] * v[i]; }
#pragma unroll
    for (int m = 16; m >= 1; m >>= 1) ss += __shfl_xor(ss, m, 32);
    float inv = 1.0f / fmaxf(sqrtf(ss), 1e-8f);

    unsigned short o[8];
#pragma unroll
    for (int i = 0; i < 8; ++i) o[i] = f2bf(v[i] * inv);
    __builtin_memcpy(ybf + (size_t)row * DIM + lane * 8, o, 16);

    const float* xr2 = x + (size_t)row * DIM + lane * 8;
#pragma unroll
    for (int i = 0; i < 8; ++i) o[i] = f2bf(xr2[i]);
    __builtin_memcpy(xbf + (size_t)row * DIM + lane * 8, o, 16);
}

// ---------------------------------------------------------------------------
// K2: Wh = x @ W, stored TRANSPOSED: whT[b][h][node] bf16 ([B][256][2048]).
// C-layout rows are consecutive nodes -> transposed store is one 16B store.
// grid (2, BN/16), 256 threads.
// ---------------------------------------------------------------------------
__global__ void k2_wh(const unsigned short* __restrict__ xbf,
                      const float* __restrict__ W,
                      unsigned short* __restrict__ whT) {
    __shared__ __align__(16) unsigned short Xi[16][256];
    __shared__ __align__(16) unsigned short Wt[128][256];   // Wt[h][k] = W[k][h0+h]
    int tid = threadIdx.x;
    int m0 = blockIdx.y * 16;
    int h0 = blockIdx.x * 128;

#pragma unroll
    for (int i = 0; i < 2; ++i) {
        int e = (i * 256 + tid) * 8;
        int m = e >> 8, k = e & 255;
        __builtin_memcpy(&Xi[m][k], xbf + (size_t)(m0 + m) * DIM + k, 16);
    }
    for (int i = 0; i < 128; ++i) {
        int idx = i * 256 + tid;
        int k = idx >> 7, h = idx & 127;
        Wt[h][k] = f2bf(W[(size_t)k * DIM + h0 + h]);
    }
    __syncthreads();

    int wv = tid >> 5, lane = tid & 31;
    F8 c;
#pragma unroll
    for (int r = 0; r < 8; ++r) c.f[r] = 0.f;
#pragma unroll
    for (int kc = 0; kc < 8; ++kc) {
        v16bf a  = load_a16(&Xi[0][0], 256, lane, kc);
        v16bf bb = load_b16(&Wt[wv * 16][0], 256, lane, kc);
        c.v = __builtin_amdgcn_wmma_f32_16x16x32_bf16(false, a, false, bb,
                                                      (short)0, c.v, false, false);
    }
    int nl = lane & 15, hf = lane >> 4;
    int h = h0 + wv * 16 + nl;
    int bb_ = m0 >> 11;                 // batch
    int node0 = (m0 & 2047) + hf * 8;   // 8 consecutive nodes per lane
    unsigned short o[8];
#pragma unroll
    for (int r = 0; r < 8; ++r) o[r] = f2bf(c.f[r]);
    __builtin_memcpy(whT + ((size_t)bb_ * DIM + h) * NNODE + node0, o, 16);
}

// ---------------------------------------------------------------------------
// K3: fused masked-softmax attention (flash style).
// grid = B * N/16, 128 threads (4 waves; wave owns 64 output features).
// Yt staged via async global->LDS; WhT tile staged via Tensor Data Mover.
// ---------------------------------------------------------------------------
__global__ void k3_attn(const unsigned short* __restrict__ ybf,
                        const unsigned short* __restrict__ whT,
                        const int* __restrict__ adj,
                        unsigned short* __restrict__ gbf) {
    __shared__ __align__(16) unsigned short Yi[16][256];
    __shared__ __align__(16) unsigned short Yt[32][256];   // B operand for scores
    __shared__ __align__(16) unsigned short WhT[256][32];  // WhT[h][m], TDM dest
    __shared__ __align__(16) unsigned short Pl[4][16][32]; // per-wave P tile

    int tid = threadIdx.x;
    int b  = blockIdx.x >> 7;
    int it = blockIdx.x & 127;
    int i0 = it * 16;
    size_t gr0 = (size_t)b * NNODE + i0;

#pragma unroll
    for (int i = 0; i < 4; ++i) {
        int e = (i * 128 + tid) * 8;
        int m = e >> 8, k = e & 255;
        __builtin_memcpy(&Yi[m][k], ybf + (gr0 + m) * DIM + k, 16);
    }
    __syncthreads();

    int wv = tid >> 5, lane = tid & 31;
    int nl = lane & 15, hf = lane >> 4;

    // Query-tile A operands are invariant across the whole j loop: keep in regs.
    v16bf areg[8];
#pragma unroll
    for (int kc = 0; kc < 8; ++kc) areg[kc] = load_a16(&Yi[0][0], 256, lane, kc);

    F8 acc[4];
#pragma unroll
    for (int t = 0; t < 4; ++t)
#pragma unroll
        for (int r = 0; r < 8; ++r) acc[t].f[r] = 0.f;
    float mrow[8], lrow[8];
#pragma unroll
    for (int r = 0; r < 8; ++r) { mrow[r] = -3.0e38f; lrow[r] = 0.f; }

    unsigned ytBase  = lds_off(&Yt[0][0]);
    unsigned whtBase = lds_off(&WhT[0][0]);

    for (int jt = 0; jt < 64; ++jt) {
        size_t gj0 = (size_t)b * NNODE + jt * 32;
        __syncthreads();   // previous iteration's Yt/WhT consumers done

        // --- TDM: 2D tile 32(nodes) x 256(features), stride 2048, 2B elems ---
        if (wv == 0) {
            unsigned long long ga =
                (unsigned long long)(uintptr_t)(whT + ((size_t)b * DIM) * NNODE + jt * 32);
            u32x4 g0 = (u32x4)0u;
            g0[0] = 1u;                                   // count = 1
            g0[1] = whtBase;                              // lds_addr
            g0[2] = (unsigned)(ga & 0xFFFFFFFFull);       // global_addr[31:0]
            g0[3] = (unsigned)((ga >> 32) & 0x1FFFFFFull) // global_addr[56:32]
                    | (2u << 30);                         // type = 2 (image)
            i32x8 g1 = (i32x8)0;
            g1[0] = (1 << 16);            // data_size = 1 (2 bytes)
            g1[1] = (int)(2048u << 16);   // tensor_dim0 = 2048 (bits 79:48)
            g1[2] = (int)(256u << 16);    // tensor_dim1 = 256 (bits 111:80)
            g1[3] = (int)(32u << 16);     // tile_dim0 = 32 (bits 127:112)
            g1[4] = 256;                  // tile_dim1 = 256, tile_dim2 = 0
            g1[5] = 2048;                 // tensor_dim0_stride = 2048
            i32x4 g2 = (i32x4)0;
            i32x4 g3 = (i32x4)0;
#if __clang_major__ >= 23
            i32x8 g4 = (i32x8)0;
            __builtin_amdgcn_tensor_load_to_lds(g0, g1, g2, g3, g4, 0);
#else
            __builtin_amdgcn_tensor_load_to_lds(g0, g1, g2, g3, 0);
#endif
        }

        // --- Async global->LDS for Yt (contiguous 16 KB) ---
        {
            const unsigned short* src = ybf + gj0 * DIM + (size_t)tid * 64;
            unsigned dst = ytBase + (unsigned)tid * 128;
#pragma unroll
            for (int i = 0; i < 8; ++i)
                async_g2l_b128(dst + i * 16, src + i * 8);
        }
        if (jt + 1 < 64) __builtin_prefetch(ybf + (gj0 + 32) * DIM, 0, 1);
        asm volatile("s_wait_asynccnt 0x0" ::: "memory");
        if (wv == 0) __builtin_amdgcn_s_wait_tensorcnt(0);
        __syncthreads();   // staged Yt/WhT visible

        // scores: two 16x16 tiles over K=256
        F8 s[2];
#pragma unroll
        for (int t = 0; t < 2; ++t) {
#pragma unroll
            for (int r = 0; r < 8; ++r) s[t].f[r] = 0.f;
#pragma unroll
            for (int kc = 0; kc < 8; ++kc) {
                v16bf bb = load_b16(&Yt[t * 16][0], 256, lane, kc);
                s[t].v = __builtin_amdgcn_wmma_f32_16x16x32_bf16(
                    false, areg[kc], false, bb, (short)0, s[t].v, false, false);
            }
        }
        // mask
        int c0 = jt * 32;
#pragma unroll
        for (int r = 0; r < 8; ++r) {
            int m = r + hf * 8;
            size_t mb = ((size_t)b * NNODE + i0 + m) * (size_t)NNODE + c0;
            if (!(adj[mb + nl] > 0))       s[0].f[r] = NEGV;
            if (!(adj[mb + 16 + nl] > 0))  s[1].f[r] = NEGV;
        }
        // online softmax (row reductions across 16-lane halves)
        float p0[8], p1[8];
#pragma unroll
        for (int r = 0; r < 8; ++r) {
            float mx = fmaxf(s[0].f[r], s[1].f[r]);
#pragma unroll
            for (int d = 8; d >= 1; d >>= 1) mx = fmaxf(mx, __shfl_xor(mx, d, 16));
            float mnew = fmaxf(mrow[r], mx);
            float scl = __expf(mrow[r] - mnew);
            p0[r] = __expf(s[0].f[r] - mnew);
            p1[r] = __expf(s[1].f[r] - mnew);
            float ps = p0[r] + p1[r];
#pragma unroll
            for (int d = 8; d >= 1; d >>= 1) ps += __shfl_xor(ps, d, 16);
            lrow[r] = lrow[r] * scl + ps;
            mrow[r] = mnew;
#pragma unroll
            for (int t = 0; t < 4; ++t) acc[t].f[r] *= scl;
        }
        // C-layout -> A-layout via LDS
#pragma unroll
        for (int r = 0; r < 8; ++r) {
            int m = r + hf * 8;
            Pl[wv][m][nl]      = f2bf(p0[r]);
            Pl[wv][m][16 + nl] = f2bf(p1[r]);
        }
        __syncthreads();
        v16bf pa = load_a16(&Pl[wv][0][0], 32, lane, 0);
#pragma unroll
        for (int t = 0; t < 4; ++t) {
            v16bf bb = load_b16(&WhT[wv * 64 + t * 16][0], 32, lane, 0);
            acc[t].v = __builtin_amdgcn_wmma_f32_16x16x32_bf16(
                false, pa, false, bb, (short)0, acc[t].v, false, false);
        }
    }

#pragma unroll
    for (int r = 0; r < 8; ++r) {
        int m = r + hf * 8;
        float inv = 1.0f / lrow[r];
#pragma unroll
        for (int t = 0; t < 4; ++t) {
            gbf[(gr0 + m) * DIM + wv * 64 + t * 16 + nl] = f2bf(acc[t].f[r] * inv);
        }
    }
}

// ---------------------------------------------------------------------------
// K4: GRU cell. grid (16, BN/16), 192 threads = 6 waves.
// ---------------------------------------------------------------------------
__global__ void k4_gru(const unsigned short* __restrict__ gbf,
                       const unsigned short* __restrict__ xbf,
                       const float* __restrict__ Wih, const float* __restrict__ Whh,
                       const float* __restrict__ bih, const float* __restrict__ bhh,
                       const float* __restrict__ x, float* __restrict__ out) {
    __shared__ float gt[6][16][16];
    int tid = threadIdx.x;
    int g = tid >> 5, lane = tid & 31;
    int h0 = blockIdx.x * 16;
    int m0 = blockIdx.y * 16;
    int nl = lane & 15, hf = lane >> 4;

    const unsigned short* A = (g < 3) ? gbf : xbf;
    const float* Wm = (g < 3) ? Wih : Whh;
    int o0 = (g % 3) * 256 + h0;     // row block of W (gate order r,z,n)

    F8 c;
#pragma unroll
    for (int r = 0; r < 8; ++r) c.f[r] = 0.f;
#pragma unroll
    for (int kc = 0; kc < 8; ++kc) {
        v16bf a = load_a16(A + (size_t)m0 * DIM, 256, lane, kc);
        BfVec u;  // B[k][n] = W[o0+n][kc*32 + k]: W rows are K-contiguous
        const float* p = Wm + (size_t)(o0 + nl) * DIM + kc * 32 + hf * 16;
#pragma unroll
        for (int e = 0; e < 16; ++e) u.s[e] = f2bf(p[e]);
        c.v = __builtin_amdgcn_wmma_f32_16x16x32_bf16(false, a, false, u.v,
                                                      (short)0, c.v, false, false);
    }
#pragma unroll
    for (int r = 0; r < 8; ++r) gt[g][r + hf * 8][nl] = c.f[r];
    __syncthreads();

    for (int idx = tid; idx < 256; idx += 192) {
        int m = idx >> 4, n = idx & 15;
        float ir = gt[0][m][n] + bih[h0 + n];
        float iz = gt[1][m][n] + bih[256 + h0 + n];
        float in_ = gt[2][m][n] + bih[512 + h0 + n];
        float hr = gt[3][m][n] + bhh[h0 + n];
        float hz = gt[4][m][n] + bhh[256 + h0 + n];
        float hn = gt[5][m][n] + bhh[512 + h0 + n];
        float rg = 1.0f / (1.0f + __expf(-(ir + hr)));
        float zg = 1.0f / (1.0f + __expf(-(iz + hz)));
        float ng = tanhf(in_ + rg * hn);
        float hv = x[(size_t)(m0 + m) * DIM + h0 + n];
        out[(size_t)(m0 + m) * DIM + h0 + n] = (1.0f - zg) * ng + zg * hv;
    }
}

// ---------------------------------------------------------------------------
extern "C" void kernel_launch(void* const* d_in, const int* in_sizes, int n_in,
                              void* d_out, int out_size, void* d_ws, size_t ws_size,
                              hipStream_t stream) {
    (void)in_sizes; (void)n_in; (void)out_size; (void)ws_size;
    const int*   adj  = (const int*)d_in[0];
    const float* x    = (const float*)d_in[1];
    const float* xt   = (const float*)d_in[2];
    const float* W    = (const float*)d_in[3];
    const float* Wih  = (const float*)d_in[4];
    const float* Whh  = (const float*)d_in[5];
    const float* bih  = (const float*)d_in[6];
    const float* bhh  = (const float*)d_in[7];
    float* out = (float*)d_out;

    unsigned short* xbf = (unsigned short*)d_ws;                // 8 MB
    unsigned short* ybf = xbf + (size_t)BN * DIM;               // 8 MB
    unsigned short* whT = ybf + (size_t)BN * DIM;               // 8 MB [B][D][N]
    unsigned short* gbf = whT + (size_t)BN * DIM;               // 8 MB

    k1_prep<<<BN / 8, 256, 0, stream>>>(x, xt, xbf, ybf);
    k2_wh  <<<dim3(2, BN / 16), 256, 0, stream>>>(xbf, W, whT);
    k3_attn<<<BATCH * (NNODE / 16), 128, 0, stream>>>(ybf, whT, adj, gbf);
    k4_gru <<<dim3(DIM / 16, BN / 16), 192, 0, stream>>>(gbf, xbf, Wih, Whh,
                                                         bih, bhh, x, out);
}
